// GraphLSTM_81492709474961
// MI455X (gfx1250) — matrix operand order repaired and checked by reading, not compiled, not run on hardware
//
#include <hip/hip_runtime.h>
#include <cstdint>

#define BB 128
#define SS 128
#define NN 24
#define FF 16
#define HH 32
#define IS (NN*FF)      /* 384  */
#define HS (NN*HH)      /* 768  */
#define G4 (4*HS)       /* 3072 */
#define NBLK 48         /* blocks in the persistent scan grid */
#define BSTRIDE 776     /* Bs row stride in halves: 1552B, 16B-aligned, bank-staggered */

typedef __attribute__((ext_vector_type(16))) __bf16 v16bf;
typedef __attribute__((ext_vector_type(8)))  float  v8f;
typedef unsigned short ushort_t;
typedef unsigned int   uint_t;

struct __align__(16) U4 { uint_t x, y, z, w; };
union FragBF { v16bf v; U4 q[2]; };

__device__ __forceinline__ ushort_t f2bf(float f) {
    uint_t u = __float_as_uint(f);
    u += 0x7FFFu + ((u >> 16) & 1u);          // round-to-nearest-even
    return (ushort_t)(u >> 16);
}

__device__ __forceinline__ float sigmoidf_(float x) {
    return 1.0f / (1.0f + __expf(-x));
}

// ---------------------------------------------------------------------------
// Wm = bf16(W * Gmask), Um = bf16(U * Gmask).
// mask element for (r,c): G[r/rowdiv][(c % HS)/HH]
// ---------------------------------------------------------------------------
__global__ void mask_cvt_kernel(const float* __restrict__ src,
                                const float* __restrict__ G,
                                ushort_t* __restrict__ dst,
                                int rows, int rowdiv) {
    int idx = blockIdx.x * blockDim.x + threadIdx.x;
    int total = rows * G4;
    if (idx >= total) return;
    int r = idx / G4, c = idx - r * G4;
    int ni = r / rowdiv;
    int nj = (c % HS) / HH;
    dst[idx] = f2bf(src[idx] * G[ni * NN + nj]);
}

__global__ void cvt_bf16_kernel(const float* __restrict__ src,
                                ushort_t* __restrict__ dst, int n) {
    int idx = blockIdx.x * blockDim.x + threadIdx.x;
    if (idx < n) dst[idx] = f2bf(src[idx]);
}

__global__ void init_hc_kernel(const float* __restrict__ h0,
                               const float* __restrict__ c0,
                               ushort_t* __restrict__ hb,
                               float* __restrict__ cb,
                               uint_t* __restrict__ barrier_cnt) {
    int idx = blockIdx.x * blockDim.x + threadIdx.x;
    if (idx == 0) *barrier_cnt = 0u;          // reset per launch (ws not re-poisoned)
    if (idx < BB * HS) { hb[idx] = f2bf(h0[idx]); cb[idx] = c0[idx]; }
}

// ---------------------------------------------------------------------------
// Phase 1: xg[16384,3072] = x_bf16[16384,384] @ Wm[384,3072] + bias
// Block: 256 threads (8 waves), tile 128(M) x 64(N), K staged 32 at a time.
// ---------------------------------------------------------------------------
__global__ __launch_bounds__(256) void gemm_xg_kernel(
    const ushort_t* __restrict__ A,   // [16384, 384] bf16
    const ushort_t* __restrict__ Bw,  // [384, 3072] bf16 (Wm)
    const float* __restrict__ bias,   // [3072]
    float* __restrict__ Cout)         // [16384, 3072] fp32
{
    __shared__ ushort_t As[128 * 40];
    __shared__ ushort_t Bs[64 * 40];
    const int t    = threadIdx.x;
    const int lane = t & 31;
    const int wv   = t >> 5;
    const int bn   = blockIdx.x;          // 0..47 (N tiles of 64)
    const long mg0 = (long)blockIdx.y * 128;

    v8f acc[4] = {};
    const int hi    = (lane >> 4) & 1;
    const int ml    = lane & 15;
    const int aoffL = (16 * wv + ml) * 40 + hi * 8;

    for (int kk = 0; kk < 12; ++kk) {
        const int k0 = kk * 32;
        #pragma unroll
        for (int rr = 0; rr < 8; ++rr) {
            int dw = t + 256 * rr;
            int m = dw >> 4, kd = dw & 15;
            uint_t val = *reinterpret_cast<const uint_t*>(A + (mg0 + m) * IS + k0 + 2 * kd);
            *reinterpret_cast<uint_t*>(&As[m * 40 + 2 * kd]) = val;
        }
        {
            int n = t & 63, kb = t >> 6;
            int cg = bn * 64 + n;
            #pragma unroll
            for (int rr = 0; rr < 8; ++rr) {
                int k = kb + 4 * rr;
                Bs[n * 40 + k] = Bw[(long)(k0 + k) * G4 + cg];
            }
        }
        __syncthreads();
        FragBF a;
        a.q[0] = *reinterpret_cast<const U4*>(&As[aoffL]);
        a.q[1] = *reinterpret_cast<const U4*>(&As[aoffL + 16]);
        #pragma unroll
        for (int nt = 0; nt < 4; ++nt) {
            FragBF b;
            int boff = (nt * 16 + ml) * 40 + hi * 8;
            b.q[0] = *reinterpret_cast<const U4*>(&Bs[boff]);
            b.q[1] = *reinterpret_cast<const U4*>(&Bs[boff + 16]);
            acc[nt] = __builtin_amdgcn_wmma_f32_16x16x32_bf16(
                false, a.v, false, b.v, (short)0, acc[nt], false, false);
        }
        __syncthreads();
    }

    const int mtile = 16 * wv;
    #pragma unroll
    for (int nt = 0; nt < 4; ++nt) {
        int col = bn * 64 + nt * 16 + ml;
        float bv = bias[col];
        #pragma unroll
        for (int r = 0; r < 8; ++r) {
            int m = mtile + r + hi * 8;
            Cout[(mg0 + m) * (long)G4 + col] = acc[nt][r] + bv;
        }
    }
}

// ---------------------------------------------------------------------------
// Phase 2: persistent scan, ONE launch for all 128 steps.
// CDNA5 320KB-LDS play: the block's full Um slice (64 gate-interleaved cols x
// 768 K, bf16, padded stride) is staged into LDS ONCE and stays resident for
// the whole scan. A-fragments are fetched straight from global h (two
// global_load_b128 per lane match the ISA 16-bit A layout), so the steady-
// state K-loop has no LDS stores and no __syncthreads at all:
//   2x global_load_b128 + 8x ds_load_b128 + 4x v_wmma per K-iter.
// Steps are separated by a software grid barrier (monotonic agent-scope
// atomic + fences). cbuf columns are block-private; only h crosses blocks.
// ---------------------------------------------------------------------------
__global__ __launch_bounds__(256) void lstm_scan_kernel(
    const float* __restrict__ xg,     // [16384, 3072]
    const ushort_t* __restrict__ Um,  // [768, 3072] bf16
    ushort_t* __restrict__ hbuf,      // [2][128*768] bf16; hbuf[0] = h0
    float* __restrict__ cbuf,         // [128, 768] fp32
    float* __restrict__ out,          // d_out
    uint_t* __restrict__ barrier_cnt)
{
    __shared__ ushort_t Bs[64 * BSTRIDE];     // 99,328 B resident Um slice
    const int t    = threadIdx.x;
    const int lane = t & 31;
    const int wv   = t >> 5;
    const int j0   = blockIdx.x * 16;         // 48 blocks cover 768 hidden cols
    const int hi    = (lane >> 4) & 1;
    const int ml    = lane & 15;
    const int j     = j0 + ml;
    const int mtile = 16 * wv;
    const int arow  = mtile + ml;             // this lane's A row (batch index)

    // ---- one-time stage of the resident B slice: Bs[n][k] = Um[k][cg(n)] ----
    {
        int n = t & 63, kb = t >> 6;          // kb in 0..3
        int cg = (n >> 4) * HS + j0 + (n & 15);   // gate-interleaved columns
        for (int rr = 0; rr < 192; ++rr) {
            int k = kb + 4 * rr;
            Bs[n * BSTRIDE + k] = Um[(long)k * G4 + cg];
        }
    }
    __syncthreads();

    for (int s = 0; s < SS; ++s) {
        const ushort_t* hin  = hbuf + (size_t)(s & 1) * (BB * HS);
        ushort_t*       hout = hbuf + (size_t)((s + 1) & 1) * (BB * HS);
        const ushort_t* arowp = hin + arow * HS + hi * 8;

        v8f acc[4] = {};
        for (int kk = 0; kk < 24; ++kk) {
            const int k0 = kk * 32;
            FragBF a;                          // A fragment straight from global
            a.q[0] = *reinterpret_cast<const U4*>(arowp + k0);
            a.q[1] = *reinterpret_cast<const U4*>(arowp + k0 + 16);
            #pragma unroll
            for (int nt = 0; nt < 4; ++nt) {
                FragBF b;
                const ushort_t* bp = &Bs[(nt * 16 + ml) * BSTRIDE + k0 + hi * 8];
                b.q[0] = *reinterpret_cast<const U4*>(bp);
                b.q[1] = *reinterpret_cast<const U4*>(bp + 16);
                acc[nt] = __builtin_amdgcn_wmma_f32_16x16x32_bf16(
                    false, a.v, false, b.v, (short)0, acc[nt], false, false);
            }
        }

        // fused LSTM pointwise; acc[0..3] = i,f,g,o partial gates for (m, j)
        #pragma unroll
        for (int r = 0; r < 8; ++r) {
            int m = mtile + r + hi * 8;                 // batch index
            long xrow = ((long)m * SS + s) * G4;
            float gi = sigmoidf_(acc[0][r] + xg[xrow + 0 * HS + j]);
            float gf = sigmoidf_(acc[1][r] + xg[xrow + 1 * HS + j]);
            float gg = tanhf   (acc[2][r] + xg[xrow + 2 * HS + j]);
            float go = sigmoidf_(acc[3][r] + xg[xrow + 3 * HS + j]);
            float cn = gf * cbuf[m * HS + j] + gi * gg;
            float hn = go * tanhf(cn);
            cbuf[m * HS + j] = cn;
            hout[m * HS + j] = f2bf(hn);
            out[((long)m * SS + s) * HS + j] = hn;      // hidden_seq[b,s,:,:]
            if (s == SS - 1) {                          // final (h, c) tail
                out[(long)BB * SS * HS + (long)m * HS + j] = hn;
                out[(long)BB * SS * HS + (long)BB * HS + (long)m * HS + j] = cn;
            }
            if (s + 1 < SS) {                           // global_prefetch_b8 next xg
                __builtin_prefetch(xg + xrow + G4 + 0 * HS + j, 0, 3);
                __builtin_prefetch(xg + xrow + G4 + 1 * HS + j, 0, 3);
                __builtin_prefetch(xg + xrow + G4 + 2 * HS + j, 0, 3);
                __builtin_prefetch(xg + xrow + G4 + 3 * HS + j, 0, 3);
            }
        }

        // ---- software grid barrier (skip after last step) ----
        if (s + 1 < SS) {
            __threadfence();                            // flush h stores to device scope
            __syncthreads();
            if (t == 0) {
                __hip_atomic_fetch_add(barrier_cnt, 1u, __ATOMIC_ACQ_REL,
                                       __HIP_MEMORY_SCOPE_AGENT);
                const uint_t target = (uint_t)NBLK * (uint_t)(s + 1);
                while (__hip_atomic_load(barrier_cnt, __ATOMIC_ACQUIRE,
                                         __HIP_MEMORY_SCOPE_AGENT) < target) {
                    __builtin_amdgcn_s_sleep(1);
                }
            }
            __syncthreads();
            __threadfence();                            // acquire: invalidate stale h lines
        }
    }
}

// ---------------------------------------------------------------------------
extern "C" void kernel_launch(void* const* d_in, const int* in_sizes, int n_in,
                              void* d_out, int out_size, void* d_ws, size_t ws_size,
                              hipStream_t stream) {
    (void)in_sizes; (void)n_in; (void)out_size; (void)ws_size;
    const float* x    = (const float*)d_in[0];   // [B,S,N,F]
    const float* h0   = (const float*)d_in[1];   // [B,N,H]
    const float* c0   = (const float*)d_in[2];   // [B,N,H]
    const float* W    = (const float*)d_in[3];   // [IS,4HS]
    const float* U    = (const float*)d_in[4];   // [HS,4HS]
    const float* bias = (const float*)d_in[5];   // [4HS]
    const float* G    = (const float*)d_in[6];   // [N,N]

    char* p = (char*)d_ws;
    float*    xg   = (float*)p;    p += (size_t)BB * SS * G4 * 4;   // 201.3 MB
    ushort_t* xbf  = (ushort_t*)p; p += (size_t)BB * SS * IS * 2;   // 12.6 MB
    ushort_t* Wm   = (ushort_t*)p; p += (size_t)IS * G4 * 2;        // 2.36 MB
    ushort_t* Um   = (ushort_t*)p; p += (size_t)HS * G4 * 2;        // 4.72 MB
    ushort_t* hbuf = (ushort_t*)p; p += (size_t)2 * BB * HS * 2;    // ping-pong h
    float*    cb   = (float*)p;    p += (size_t)BB * HS * 4;
    uint_t*   cnt  = (uint_t*)p;   p += 128;                        // barrier counter

    float* out = (float*)d_out;

    mask_cvt_kernel<<<(IS * G4 + 255) / 256, 256, 0, stream>>>(W, G, Wm, IS, FF);
    mask_cvt_kernel<<<(HS * G4 + 255) / 256, 256, 0, stream>>>(U, G, Um, HS, HH);
    cvt_bf16_kernel<<<(BB * SS * IS + 255) / 256, 256, 0, stream>>>(x, xbf, BB * SS * IS);
    init_hc_kernel<<<(BB * HS + 255) / 256, 256, 0, stream>>>(h0, c0, hbuf, cb, cnt);

    gemm_xg_kernel<<<dim3(48, 128), 256, 0, stream>>>(xbf, Wm, bias, xg);

    lstm_scan_kernel<<<NBLK, 256, 0, stream>>>(xg, Um, hbuf, cb, out, cnt);
}
